// kMaXTransformerLayer_1039382085851
// MI455X (gfx1250) — compile-verified, matrix operand order checked by reading, not run on hardware
//
#include <hip/hip_runtime.h>
#include <hip/hip_bf16.h>
#include <math.h>

typedef _Float16 h16 __attribute__((ext_vector_type(16)));
typedef _Float16 h8t __attribute__((ext_vector_type(8)));
typedef float    f8t __attribute__((ext_vector_type(8)));

union Frag16 { h16 v; h8t h[2]; };

__device__ __forceinline__ float geluf(float x) {
  // exact gelu: 0.5*x*(1+erf(x/sqrt(2)))
  return 0.5f * x * (1.0f + erff(x * 0.70710678118654752f));
}

#define LDT 48          // LDS tile K-stride in f16 (96B rows -> 16B aligned b128 reads)
#define MPB 8192        // spatial size per batch (32*32*8)

// =====================================================================
// WMMA GEMM:  Out[b][o][m] = sum_k A(o,k;b) * T(B[b][k][m]) (+ bias[o])
//   B layout [Bn][K][8192] fp32.  tmode: 0=identity, 1=gelu, 2=bn(scale/shift)+gelu
//   A element (o,k) at A + b*a_bs + o*a_os + k*a_ks  (a_bs=0 -> shared weights)
// Block: 128 threads (4 waves). Tile: 32 (O) x 256 (M). K chunked by 32.
// Per wave per chunk: 2 A frags + 4 B frags (b128 LDS reads) -> 8 WMMAs.
// Cooperative loads use a single affine pointer per thread (no spill pressure).
// =====================================================================
__global__ void wmma_gemm_k(const float* __restrict__ A, const float* __restrict__ Bm,
                            float* __restrict__ Out,
                            const float* __restrict__ iSc, const float* __restrict__ iSh,
                            const float* __restrict__ bias,
                            int O, int K, int tmode, int a_os, int a_ks, int a_bs)
{
  __shared__ __align__(16) _Float16 At[32 * LDT];
  __shared__ __align__(16) _Float16 Bt[256 * LDT];

  const int t    = threadIdx.x;
  const int col0 = blockIdx.x * 256;           // global column tile
  const int b    = col0 >> 13;                 // batch (8192 cols per batch, tiles never cross)
  const int m0   = col0 & (MPB - 1);
  const int o0   = blockIdx.y * 32;

  const int lane = t & 31;
  const int wv   = t >> 5;
  const int r    = lane & 15;
  const int half = lane >> 4;

  f8t acc[2][4] = {};

  // ---- per-thread affine load coordinates ----
  const int bn   = (t & 63) * 4;   // B: column within tile (4 consecutive)
  const int bk   = t >> 6;         // B: starting k row (0/1), step 2, 16 steps
  const int ak1  = (t & 7) * 4;    // A fast path: k within chunk (4 consecutive)
  const int ao1  = t >> 3;         // A fast path: o row (0..15), +16 on 2nd iter
  const int akg  = t & 31;         // A generic: k within chunk
  const int aog  = t >> 5;         // A generic: o row (0..3), step 4, 8 steps

  const float* Bbase = Bm + (long)b * K * MPB + m0;
  const float* Abase = A + (long)b * a_bs;

  // single pointer induction variables (advance chunk-to-chunk)
  const float* pB  = Bbase + (long)bk * MPB + bn;                    // steps 2*MPB x16 = 32*MPB
  const float* pA1 = Abase + (long)(o0 + ao1) * a_os + ak1;          // reset via +32 per chunk
  const float* pAg = Abase + (long)(o0 + aog) * a_os + (long)akg * a_ks;

  for (int k0 = 0; k0 < K; k0 += 32) {
    // ---- cooperative A tile load: 32 x 32 ----
    if (a_ks == 1) {
      const float* pa = pA1;
      #pragma unroll
      for (int i = 0; i < 2; ++i) {
        float4 v = *(const float4*)pa;
        _Float16* ats = &At[(ao1 + 16 * i) * LDT + ak1];
        ats[0] = (_Float16)v.x; ats[1] = (_Float16)v.y;
        ats[2] = (_Float16)v.z; ats[3] = (_Float16)v.w;
        pa += (long)16 * a_os;
      }
      pA1 += 32;
    } else {                                    // generic strided A (mask-logits cmk)
      const float* pa = pAg;
      _Float16* ats = &At[aog * LDT + akg];
      #pragma unroll
      for (int i = 0; i < 8; ++i) {
        *ats = (_Float16)(*pa);
        pa  += (long)4 * a_os;
        ats += 4 * LDT;
      }
      pAg += 32 * a_ks;
    }
    // ---- cooperative B tile load (float4, transposed to [n][k]) + fused transform ----
    {
      _Float16* bts = &Bt[bn * LDT + bk];
      #pragma unroll 4
      for (int i = 0; i < 16; ++i) {
        float4 v = *(const float4*)pB;
        int kk = k0 + bk + 2 * i;
        float s = 1.f, d = 0.f;
        if (tmode == 2) { s = iSc[kk]; d = iSh[kk]; }
        float x0 = v.x * s + d, x1 = v.y * s + d, x2 = v.z * s + d, x3 = v.w * s + d;
        if (tmode >= 1) { x0 = geluf(x0); x1 = geluf(x1); x2 = geluf(x2); x3 = geluf(x3); }
        bts[0]       = (_Float16)x0;
        bts[LDT]     = (_Float16)x1;
        bts[2 * LDT] = (_Float16)x2;
        bts[3 * LDT] = (_Float16)x3;
        pB  += 2 * MPB;      // after 16 iters -> advanced exactly one K chunk
        bts += 2;
      }
    }
    __syncthreads();

    // ---- fragments per the CDNA5 16-bit layouts ----
    Frag16 af[2];
    #pragma unroll
    for (int p = 0; p < 2; ++p) {
      af[p].h[0] = *(const h8t*)&At[(p * 16 + r) * LDT + half * 8];
      af[p].h[1] = *(const h8t*)&At[(p * 16 + r) * LDT + 16 + half * 8];
    }
    #pragma unroll
    for (int j = 0; j < 4; ++j) {
      int n = wv * 64 + j * 16 + r;
      Frag16 bf;
      bf.h[0] = *(const h8t*)&Bt[n * LDT + half * 8];
      bf.h[1] = *(const h8t*)&Bt[n * LDT + 16 + half * 8];
      #pragma unroll
      for (int p = 0; p < 2; ++p)
        acc[p][j] = __builtin_amdgcn_wmma_f32_16x16x32_f16(
            false, af[p].v, false, bf.v, (short)0, acc[p][j], false, false);
    }
    __syncthreads();
  }

  // ---- store: C/D layout VGPR rr -> row (half?8:0)+rr, col = lane&15 ----
  #pragma unroll
  for (int p = 0; p < 2; ++p)
  #pragma unroll
  for (int j = 0; j < 4; ++j) {
    int n = wv * 64 + j * 16 + r;
    #pragma unroll
    for (int rr = 0; rr < 8; ++rr) {
      int orow = o0 + p * 16 + half * 8 + rr;
      float v = acc[p][j][rr];
      if (bias) v += bias[orow];
      Out[(long)b * O * MPB + (long)orow * MPB + m0 + n] = v;
    }
  }
}

// =====================================================================
// Per-channel training-mode BN stats -> (scale, shift).  X[b][o][M]
// =====================================================================
__global__ void channel_stats_k(const float* __restrict__ X, float* sc, float* sh,
                                const float* __restrict__ g, const float* __restrict__ bt,
                                int O, int M, int Bn)
{
  int o = blockIdx.x, t = threadIdx.x;
  __shared__ float s1[256], s2[256];
  float a = 0.f, q = 0.f;
  int tot = M * Bn;
  for (int i = t; i < tot; i += 256) {
    int bb = i / M, m = i - bb * M;
    float v = X[(long)bb * O * M + (long)o * M + m];
    a += v; q += v * v;
  }
  s1[t] = a; s2[t] = q; __syncthreads();
  for (int st = 128; st > 0; st >>= 1) {
    if (t < st) { s1[t] += s1[t + st]; s2[t] += s2[t + st]; }
    __syncthreads();
  }
  if (t == 0) {
    float mean = s1[0] / tot;
    float var  = s2[0] / tot - mean * mean;
    float scl  = g[o] * rsqrtf(fmaxf(var, 0.f) + 1e-5f);
    sc[o] = scl; sh[o] = bt[o] - mean * scl;
  }
}

// =====================================================================
// Depthwise 5x5x5 conv (pad 2) on bn-transformed input. block=(w,s)=256
// =====================================================================
__global__ void dwconv_k(const float* __restrict__ X, const float* __restrict__ sc,
                         const float* __restrict__ sh, const float* __restrict__ W5,
                         float* __restrict__ Out)
{
  int t = threadIdx.x;
  int h = blockIdx.x & 31;
  int c = (blockIdx.x >> 5) & 255;
  int b = blockIdx.x >> 13;
  __shared__ float wl[125];
  __shared__ float ssc[2];
  if (t < 125) wl[t] = W5[c * 125 + t];
  if (t == 128) ssc[0] = sc[c];
  if (t == 129) ssc[1] = sh[c];
  __syncthreads();
  int w = t >> 3, s = t & 7;
  const float* Xc = X + ((long)b * 256 + c) * MPB;
  float acc = 0.f;
  for (int dh = 0; dh < 5; ++dh) {
    int hh = h + dh - 2; if (hh < 0 || hh > 31) continue;
    for (int dw = 0; dw < 5; ++dw) {
      int ww = w + dw - 2; if (ww < 0 || ww > 31) continue;
      for (int ds = 0; ds < 5; ++ds) {
        int ss = s + ds - 2; if (ss < 0 || ss > 7) continue;
        acc += wl[(dh * 5 + dw) * 5 + ds] * (Xc[hh * 256 + ww * 8 + ss] * ssc[0] + ssc[1]);
      }
    }
  }
  Out[((long)b * 256 + c) * MPB + h * 256 + w * 8 + s] = acc;
}

// =====================================================================
// pix_norm: apply bn then L2-normalize across 128 channels, per column.
// =====================================================================
__global__ void pixnorm_k(const float* __restrict__ X, const float* __restrict__ sc,
                          const float* __restrict__ sh, float* __restrict__ Out)
{
  int i = blockIdx.x * 256 + threadIdx.x;
  if (i >= 2 * MPB) return;
  int b = i >> 13, m = i & (MPB - 1);
  const float* p = X + (long)b * 128 * MPB + m;
  float ss = 0.f;
  for (int c = 0; c < 128; ++c) { float v = p[c * MPB] * sc[c] + sh[c]; ss += v * v; }
  float inv = 1.0f / fmaxf(sqrtf(ss), 1e-12f);
  float* o = Out + (long)b * 128 * MPB + m;
  for (int c = 0; c < 128; ++c) o[c * MPB] = (p[c * MPB] * sc[c] + sh[c]) * inv;
}

// =====================================================================
// argmax over 64 cluster logits (bn applied) per column
// =====================================================================
__global__ void argmax_k(const float* __restrict__ ML, const float* __restrict__ sc,
                         const float* __restrict__ sh, int* __restrict__ idx)
{
  int i = blockIdx.x * 256 + threadIdx.x;
  if (i >= 2 * MPB) return;
  int b = i >> 13, m = i & (MPB - 1);
  const float* p = ML + (long)b * 64 * MPB + m;
  float best = -3.4e38f; int bi = 0;
  for (int n = 0; n < 64; ++n) {
    float v = p[n * MPB] * sc[n] + sh[n];
    if (v > best) { best = v; bi = n; }
  }
  idx[i] = bi;
}

__global__ void zero_k(float* p, int n) {
  int i = blockIdx.x * 256 + threadIdx.x;
  if (i < n) p[i] = 0.f;
}

// =====================================================================
// kmu scatter: kmu[b][d][l] += bn(pixel_value[b][d][m]) for idx[b,m]==l
// grid(32, 2), block 128; LDS accumulation, one atomic flush per (d,l).
// =====================================================================
__global__ void scatter_kmu_k(const float* __restrict__ PV, const float* __restrict__ sc,
                              const float* __restrict__ sh, const int* __restrict__ idx,
                              float* __restrict__ kmu)
{
  __shared__ float lacc[64 * 128];
  int t = threadIdx.x;
  int d = blockIdx.y * 128 + t;
  for (int l = 0; l < 64; ++l) lacc[l * 128 + t] = 0.f;
  int col0 = blockIdx.x * 512;
  int b = col0 >> 13, m0 = col0 & (MPB - 1);
  float scv = sc[d], shv = sh[d];
  const float* pvb = PV + ((long)b * 256 + d) * MPB + m0;
  const int* ib = idx + col0;
  for (int j = 0; j < 512; ++j) {
    int l = ib[j];
    lacc[l * 128 + t] += pvb[j] * scv + shv;
  }
  float* kb = kmu + (long)b * 256 * 64 + (long)d * 64;
  for (int l = 0; l < 64; ++l) atomicAdd(&kb[l], lacc[l * 128 + t]);
}

// =====================================================================
// Query-side small ops (L = 64)
// =====================================================================
__global__ void small_gemm_k(const float* __restrict__ W, const float* __restrict__ X,
                             float* __restrict__ Y, int O, int K, int Bn)
{
  int i = blockIdx.x * 256 + threadIdx.x;
  if (i >= Bn * O * 64) return;
  int l = i & 63, o = (i >> 6) % O, b = i / (O * 64);
  const float* x = X + (long)b * K * 64;
  const float* w = W + (long)o * K;
  float acc = 0.f;
  for (int k = 0; k < K; ++k) acc += w[k] * x[k * 64 + l];
  Y[i] = acc;
}

// y = act(x*sc[ch]+sh[ch] (+res))   ch = (i>>6)%O ; act: 0 none, 1 gelu
__global__ void apply_bn_k(const float* __restrict__ X, const float* __restrict__ sc,
                           const float* __restrict__ sh, const float* __restrict__ res,
                           float* __restrict__ Y, int O, int total, int act)
{
  int i = blockIdx.x * 256 + threadIdx.x;
  if (i >= total) return;
  int o = (i >> 6) % O;
  float v = X[i] * sc[o] + sh[o];
  if (res) v += res[i];
  if (act) v = geluf(v);
  Y[i] = v;
}

__global__ void add_voidbias_k(const float* __restrict__ X, float* __restrict__ Y, int total)
{
  int i = blockIdx.x * 256 + threadIdx.x;
  if (i >= total) return;
  float v = X[i];
  if ((i & 63) == 63) v += logf(567.0f);   // log((L-1)*0.9/0.1), L=64
  Y[i] = v;
}

__global__ void sim_qk_k(const float* __restrict__ QKV, float* __restrict__ S)
{
  int i = blockIdx.x * 256 + threadIdx.x;   // 2*8*64*64 = 65536
  if (i >= 65536) return;
  int m = i & 63, l = (i >> 6) & 63, h = (i >> 12) & 7, b = i >> 15;
  const float* q = QKV + (long)b * 512 * 64 + (h * 16) * 64 + l;
  const float* k = QKV + (long)b * 512 * 64 + (128 + h * 16) * 64 + m;
  float acc = 0.f;
  for (int d = 0; d < 16; ++d) acc += q[d * 64] * k[d * 64];
  S[i] = acc;
}

__global__ void softmax_bn_k(const float* __restrict__ S, const float* __restrict__ sc,
                             const float* __restrict__ sh, float* __restrict__ A)
{
  int i = blockIdx.x * 256 + threadIdx.x;   // 2*8*64 = 1024 rows
  if (i >= 1024) return;
  int l = i & 63, h = (i >> 6) & 7, b = i >> 9;
  long base = ((long)(b * 8 + h) * 64 + l) * 64;
  const float* row = S + base;
  float* arow = A + base;
  float scv = sc[h], shv = sh[h];
  float mx = -3.4e38f;
  for (int m = 0; m < 64; ++m) mx = fmaxf(mx, row[m] * scv + shv);
  float sum = 0.f;
  for (int m = 0; m < 64; ++m) { float e = expf(row[m] * scv + shv - mx); arow[m] = e; sum += e; }
  float inv = 1.0f / sum;
  for (int m = 0; m < 64; ++m) arow[m] *= inv;
}

__global__ void attn_v_k(const float* __restrict__ AW, const float* __restrict__ QKV,
                         float* __restrict__ RV)
{
  int i = blockIdx.x * 256 + threadIdx.x;   // 2*256*64 = 32768
  if (i >= 32768) return;
  int l = i & 63, c = (i >> 6) & 255, b = i >> 14;
  int h = c >> 5;
  const float* a = AW + ((long)(b * 8 + h) * 64 + l) * 64;
  const float* v = QKV + (long)b * 512 * 64 + (256 + c) * 64;
  float acc = 0.f;
  for (int m = 0; m < 64; ++m) acc += a[m] * v[m];
  RV[i] = acc;   // layout [b][256][64]
}

// =====================================================================
extern "C" void kernel_launch(void* const* d_in, const int* in_sizes, int n_in,
                              void* d_out, int out_size, void* d_ws, size_t ws_size,
                              hipStream_t stream)
{
  const float* pfK = (const float*)d_in[0];
  const float* pfV = (const float*)d_in[1];
  const float* qf0 = (const float*)d_in[2];
  const float *q1_w=(const float*)d_in[3], *q1_g=(const float*)d_in[4], *q1_b=(const float*)d_in[5];
  const float *q2_w=(const float*)d_in[6], *q2_g=(const float*)d_in[7], *q2_b=(const float*)d_in[8];
  const float *pvc_w=(const float*)d_in[9], *pvc_g=(const float*)d_in[10], *pvc_b=(const float*)d_in[11];
  const float *pkc_w=(const float*)d_in[12],*pkc_g=(const float*)d_in[13],*pkc_b=(const float*)d_in[14];
  const float *qkv_w=(const float*)d_in[15],*qkv_g=(const float*)d_in[16],*qkv_b=(const float*)d_in[17];
  const float *pv_w=(const float*)d_in[18], *pv_g=(const float*)d_in[19], *pv_b=(const float*)d_in[20];
  const float *pk_w=(const float*)d_in[21], *pk_g=(const float*)d_in[22], *pk_b=(const float*)d_in[23];
  const float *sim_g=(const float*)d_in[24],*sim_b=(const float*)d_in[25];
  const float *rv_g=(const float*)d_in[26], *rv_b=(const float*)d_in[27];
  const float *q3_w=(const float*)d_in[28], *q3_g=(const float*)d_in[29], *q3_b=(const float*)d_in[30];
  const float *f1_w=(const float*)d_in[31], *f1_g=(const float*)d_in[32], *f1_b=(const float*)d_in[33];
  const float *f2_w=(const float*)d_in[34], *f2_g=(const float*)d_in[35], *f2_b=(const float*)d_in[36];
  const float *krv_g=(const float*)d_in[37],*krv_b=(const float*)d_in[38];
  const float *kq3_w=(const float*)d_in[39],*kq3_g=(const float*)d_in[40],*kq3_b=(const float*)d_in[41];
  const float *dw_w=(const float*)d_in[42], *dw_g=(const float*)d_in[43], *dw_b=(const float*)d_in[44];
  const float *h1_w=(const float*)d_in[45], *h1_g=(const float*)d_in[46], *h1_b=(const float*)d_in[47];
  const float *hl_w=(const float*)d_in[48], *hl_bias=(const float*)d_in[49],
              *hl_g=(const float*)d_in[50], *hl_b=(const float*)d_in[51];
  const float *mh_w=(const float*)d_in[52], *mh_g=(const float*)d_in[53], *mh_b=(const float*)d_in[54];
  const float *mbn_g=(const float*)d_in[55],*mbn_b=(const float*)d_in[56];

  float* out_qf = (float*)d_out;            // [2,16,64]
  float* out_pn = (float*)d_out + 2048;     // [2,128,8192]

  // ---- bump allocator over d_ws ----
  char* wp = (char*)d_ws;
  auto alloc = [&](size_t nfloat) -> float* {
    float* p = (float*)wp;
    wp += ((nfloat * 4 + 255) / 256) * 256;
    return p;
  };
  float* bufA = alloc(256 * 16384);  // 16 MB (V1 / dw-out / mask-logits)
  float* bufB = alloc(256 * 16384);  // 16 MB (PVraw, long-lived)
  float* bufC = alloc(256 * 16384);  // 16 MB (K1 / h1-out)
  float* bufD = alloc(256 * 16384);  // 16 MB (K2 / hl-out)

  auto allocSS = [&](int O, float** sc, float** sh) { *sc = alloc(O); *sh = alloc(O); };
  float *sc_v1,*sh_v1,*sc_v2,*sh_v2,*sc_k1,*sh_k1,*sc_k2,*sh_k2,*sc_dw,*sh_dw,*sc_h1,*sh_h1,
        *sc_hl,*sh_hl,*sc_ml,*sh_ml,*sc_q1,*sh_q1,*sc_mh,*sh_mh,*sc_kr,*sh_kr,*sc_k3,*sh_k3,
        *sc_q2,*sh_q2,*sc_qk,*sh_qk,*sc_si,*sh_si,*sc_rv,*sh_rv,*sc_q3,*sh_q3,*sc_f1,*sh_f1,
        *sc_f2,*sh_f2;
  allocSS(256,&sc_v1,&sh_v1); allocSS(256,&sc_v2,&sh_v2); allocSS(256,&sc_k1,&sh_k1);
  allocSS(256,&sc_k2,&sh_k2); allocSS(256,&sc_dw,&sh_dw); allocSS(256,&sc_h1,&sh_h1);
  allocSS(128,&sc_hl,&sh_hl); allocSS(64,&sc_ml,&sh_ml);  allocSS(256,&sc_q1,&sh_q1);
  allocSS(128,&sc_mh,&sh_mh); allocSS(256,&sc_kr,&sh_kr); allocSS(16,&sc_k3,&sh_k3);
  allocSS(256,&sc_q2,&sh_q2); allocSS(512,&sc_qk,&sh_qk); allocSS(8,&sc_si,&sh_si);
  allocSS(256,&sc_rv,&sh_rv); allocSS(16,&sc_q3,&sh_q3);  allocSS(2048,&sc_f1,&sh_f1);
  allocSS(16,&sc_f2,&sh_f2);

  int*   idx    = (int*)alloc(2 * MPB);
  float* kmu    = alloc(2 * 256 * 64);
  float* qs     = alloc(2 * 256 * 64);
  float* cmki   = alloc(2 * 256 * 64);
  float* cmkraw = alloc(2 * 128 * 64);
  float* cmk    = alloc(2 * 128 * 64);
  float* kmub   = alloc(2 * 256 * 64);
  float* t16    = alloc(2 * 16 * 64);
  float* qf1    = alloc(2 * 16 * 64);
  float* t256   = alloc(2 * 256 * 64);
  float* qs2    = alloc(2 * 256 * 64);
  float* qkvraw = alloc(2 * 512 * 64);
  float* qkvb   = alloc(2 * 512 * 64);
  float* simraw = alloc(2 * 8 * 64 * 64);
  float* aw     = alloc(2 * 8 * 64 * 64);
  float* rvraw  = alloc(2 * 256 * 64);
  float* rvb    = alloc(2 * 256 * 64);
  float* qf2    = alloc(2 * 16 * 64);
  float* ffraw  = alloc(2 * 2048 * 64);
  float* ffb    = alloc(2 * 2048 * 64);

  auto gemm = [&](const float* A, const float* B, float* Out, const float* isc,
                  const float* ish, const float* bias, int O, int K, int tmode,
                  int a_os, int a_ks, int a_bs) {
    wmma_gemm_k<<<dim3(64, O / 32), dim3(128), 0, stream>>>(A, B, Out, isc, ish, bias,
                                                            O, K, tmode, a_os, a_ks, a_bs);
  };
  auto stats = [&](const float* X, int O, int M, int Bn, const float* g, const float* bt,
                   float* sc, float* sh) {
    channel_stats_k<<<dim3(O), dim3(256), 0, stream>>>(X, sc, sh, g, bt, O, M, Bn);
  };
  auto sgemm = [&](const float* W, const float* X, float* Y, int O, int K) {
    int tot = 2 * O * 64;
    small_gemm_k<<<dim3((tot + 255) / 256), dim3(256), 0, stream>>>(W, X, Y, O, K, 2);
  };
  auto apply = [&](const float* X, const float* sc, const float* sh, const float* res,
                   float* Y, int O, int act) {
    int tot = 2 * O * 64;
    apply_bn_k<<<dim3((tot + 255) / 256), dim3(256), 0, stream>>>(X, sc, sh, res, Y, O, tot, act);
  };

  // ================= pixel V path =================
  gemm(pvc_w, pfV, bufA, nullptr, nullptr, nullptr, 256, 2048, /*gelu*/1, 2048, 1, 0);
  stats(bufA, 256, MPB, 2, pvc_g, pvc_b, sc_v1, sh_v1);
  gemm(pv_w, bufA, bufB, sc_v1, sh_v1, nullptr, 256, 256, /*bn+gelu*/2, 256, 1, 0);
  stats(bufB, 256, MPB, 2, pv_g, pv_b, sc_v2, sh_v2);   // pixel_value = bn(bufB) fused later

  // ================= pixel K path =================
  gemm(pkc_w, pfK, bufC, nullptr, nullptr, nullptr, 256, 2048, 1, 2048, 1, 0);
  stats(bufC, 256, MPB, 2, pkc_g, pkc_b, sc_k1, sh_k1);
  gemm(pk_w, bufC, bufD, sc_k1, sh_k1, nullptr, 256, 256, 2, 256, 1, 0);
  stats(bufD, 256, MPB, 2, pk_g, pk_b, sc_k2, sh_k2);   // pixel_key = bn(bufD) fused into dw

  // depthwise 5x5x5 conv (bn fused on load)
  dwconv_k<<<dim3(2 * 256 * 32), dim3(256), 0, stream>>>(bufD, sc_k2, sh_k2, dw_w, bufA);
  stats(bufA, 256, MPB, 2, dw_g, dw_b, sc_dw, sh_dw);

  gemm(h1_w, bufA, bufC, sc_dw, sh_dw, nullptr, 256, 256, 2, 256, 1, 0);
  stats(bufC, 256, MPB, 2, h1_g, h1_b, sc_h1, sh_h1);

  gemm(hl_w, bufC, bufD, sc_h1, sh_h1, hl_bias, 128, 256, 2, 256, 1, 0);
  stats(bufD, 128, MPB, 2, hl_g, hl_b, sc_hl, sh_hl);

  pixnorm_k<<<dim3(64), dim3(256), 0, stream>>>(bufD, sc_hl, sh_hl, out_pn);

  // ================= query: q1 -> cmk =================
  sgemm(q1_w, qf0, qs, 256, 16);
  stats(qs, 256, 64, 2, q1_g, q1_b, sc_q1, sh_q1);
  apply(qs, sc_q1, sh_q1, nullptr, qs, 256, 1);                  // query_space
  add_voidbias_k<<<dim3(128), dim3(256), 0, stream>>>(qs, cmki, 2 * 256 * 64);
  sgemm(mh_w, cmki, cmkraw, 128, 256);
  stats(cmkraw, 128, 64, 2, mh_g, mh_b, sc_mh, sh_mh);
  apply(cmkraw, sc_mh, sh_mh, nullptr, cmk, 128, 0);             // cmk [2,128,64]

  // mask_logits = cmk^T x pix_norm  (per-batch A)
  gemm(cmk, out_pn, bufA, nullptr, nullptr, nullptr, 64, 128, 0, /*a_os*/1, /*a_ks*/64, /*a_bs*/8192);
  stats(bufA, 64, MPB, 2, mbn_g, mbn_b, sc_ml, sh_ml);

  // hard cluster assignment + kmu scatter (pixel_value bn fused)
  argmax_k<<<dim3(64), dim3(256), 0, stream>>>(bufA, sc_ml, sh_ml, idx);
  zero_k<<<dim3(128), dim3(256), 0, stream>>>(kmu, 2 * 256 * 64);
  scatter_kmu_k<<<dim3(32, 2), dim3(128), 0, stream>>>(bufB, sc_v2, sh_v2, idx, kmu);

  // kmu -> bn -> kq3 -> bn -> residual add
  stats(kmu, 256, 64, 2, krv_g, krv_b, sc_kr, sh_kr);
  apply(kmu, sc_kr, sh_kr, nullptr, kmub, 256, 0);
  sgemm(kq3_w, kmub, t16, 16, 256);
  stats(t16, 16, 64, 2, kq3_g, kq3_b, sc_k3, sh_k3);
  apply(t16, sc_k3, sh_k3, qf0, qf1, 16, 0);                     // qf1 = qf0 + bn(kq3)

  // ================= query self-attention =================
  sgemm(q2_w, qf1, t256, 256, 16);
  stats(t256, 256, 64, 2, q2_g, q2_b, sc_q2, sh_q2);
  apply(t256, sc_q2, sh_q2, nullptr, qs2, 256, 1);
  sgemm(qkv_w, qs2, qkvraw, 512, 256);
  stats(qkvraw, 512, 64, 2, qkv_g, qkv_b, sc_qk, sh_qk);
  apply(qkvraw, sc_qk, sh_qk, nullptr, qkvb, 512, 0);
  sim_qk_k<<<dim3(256), dim3(256), 0, stream>>>(qkvb, simraw);
  stats(simraw, 8, 4096, 2, sim_g, sim_b, sc_si, sh_si);
  softmax_bn_k<<<dim3(4), dim3(256), 0, stream>>>(simraw, sc_si, sh_si, aw);
  attn_v_k<<<dim3(128), dim3(256), 0, stream>>>(aw, qkvb, rvraw);
  stats(rvraw, 256, 64, 2, rv_g, rv_b, sc_rv, sh_rv);
  apply(rvraw, sc_rv, sh_rv, nullptr, rvb, 256, 1);
  sgemm(q3_w, rvb, t16, 16, 256);
  stats(t16, 16, 64, 2, q3_g, q3_b, sc_q3, sh_q3);
  apply(t16, sc_q3, sh_q3, qf1, qf2, 16, 1);                     // qf2 = gelu(qf1 + upd)

  // ================= FFN =================
  sgemm(f1_w, qf2, ffraw, 2048, 16);
  stats(ffraw, 2048, 64, 2, f1_g, f1_b, sc_f1, sh_f1);
  apply(ffraw, sc_f1, sh_f1, nullptr, ffb, 2048, 1);
  sgemm(f2_w, ffb, t16, 16, 2048);
  stats(t16, 16, 64, 2, f2_g, f2_b, sc_f2, sh_f2);
  apply(t16, sc_f2, sh_f2, qf2, out_qf, 16, 1);                  // final query_feature
}
